// CTRNN_18571438588105
// MI455X (gfx1250) — matrix operand order repaired and checked by reading, not compiled, not run on hardware
//
#include <hip/hip_runtime.h>
#include <utility>

// ---------------------------------------------------------------------------
// CTRNN forward on gfx1250 (MI455X), wave32 + WMMA bf16.
//
//   T=512, B=128, I=64, H=512, O=32, alpha=0.1
//   u      = x_t @ W_in^T + b_in              (fused per step)
//   v      = (1-a) v + a (u + fr @ W_h^T + b_h)
//   fr     = relu(v)              -> hidden_seq (T,B,H)  [output part 2]
//   logits = hidden @ W_out^T + b_out          (T,B,O)   [output part 1]
//
// Recurrence: ONE persistent 512-thread workgroup (16 waves, 4/SIMD).
//   - fr / x in LDS pre-swizzled into the WMMA A-operand tile layout
//     ([kt][mt][lane][32B]) -> immediate-offset ds accesses.
//   - v state lives in the WMMA f32 accumulators (alpha folded into packed
//     weights:  acc = (1-a)*acc + (aW_in)x + (aW_h)fr + a*bias)
//   - W_in(2) + W_h(16) K-tiles packed per N-strip as one B stream; streamed
//     from L2 every step via inline-asm global_load_b128 with COMPILE-TIME
//     24-bit immediate offsets (one base pointer, no pointer-array pressure),
//     double-buffered with explicit s_wait_loadcnt.
//   - A tiles ping-pong so ds_load latency overlaps the WMMA pair.
//   - Each wave owns 2 adjacent N-strips: each A tile feeds 2 WMMAs.
// ---------------------------------------------------------------------------

#define T_STEPS 512
#define BATCH   128
#define IN_DIM  64
#define HID     512
#define OUT_DIM 32

typedef __attribute__((ext_vector_type(16))) __bf16       v16bf;
typedef __attribute__((ext_vector_type(8)))  __bf16       v8bf;
typedef __attribute__((ext_vector_type(8)))  float        v8f;
typedef __attribute__((ext_vector_type(4)))  float        v4f;
typedef __attribute__((ext_vector_type(4)))  unsigned int v4u;

constexpr float K_ALPHA   = 0.1f;
constexpr int   N_STAGES  = 18;        // 2 W_in K-tiles + 16 W_h K-tiles
constexpr int   TILE_ELEM = 512;       // 32 lanes * 16 bf16 per 16x32/32x16 tile
constexpr int   STRIP     = N_STAGES * TILE_ELEM;  // 9216 elems per N-strip
constexpr int   STRIP_B   = STRIP * 2;             // 18432 bytes per N-strip

// compile-time for-loop (drives immediate asm offsets)
template <typename F, int... Is>
__device__ __forceinline__ void static_for_impl(F&& f,
                                                std::integer_sequence<int, Is...>) {
  (f(std::integral_constant<int, Is>{}), ...);
}
template <int N, typename F>
__device__ __forceinline__ void static_for(F&& f) {
  static_for_impl(f, std::make_integer_sequence<int, N>{});
}

// ---------------------------------------------------------------------------
// WMMA wrapper: D = A(16x32 bf16) * B(32x16 bf16) + C(16x16 f32)
// ---------------------------------------------------------------------------
__device__ __forceinline__ v8f wmma_bf16(v16bf a, v16bf b, v8f c) {
  return __builtin_amdgcn_wmma_f32_16x16x32_bf16(
      /*neg_a=*/false, a, /*neg_b=*/false, b,
      /*c_mod=*/(short)0, c, /*reuse_a=*/false, /*reuse_b=*/false);
}

__device__ __forceinline__ v16bf combine_b(v4u lo, v4u hi) {
  union { v4u u[2]; v16bf b; } c;
  c.u[0] = lo;
  c.u[1] = hi;
  return c.b;
}

// ---------------------------------------------------------------------------
// B-operand global load with compile-time immediate offset, issued via inline
// asm so it is re-executed every timestep (never hoisted/spilled).  One base
// pointer serves all 36 loads of a step.  Vector loads complete in order:
// with the next stage's 4 loads in flight, "loadcnt <= 4" guarantees the
// current stage's 4 loads have landed.
// ---------------------------------------------------------------------------
template <int BYTE_OFF>
__device__ __forceinline__ void load_b_imm(const __bf16* p, v4u& lo, v4u& hi) {
  asm volatile("global_load_b128 %0, %2, off offset:%3\n\t"
               "global_load_b128 %1, %2, off offset:%4"
               : "=&v"(lo), "=&v"(hi)
               : "v"(p), "i"(BYTE_OFF), "i"(BYTE_OFF + 16)
               : "memory");
}
__device__ __forceinline__ void wait_ld0() {
  asm volatile("s_wait_loadcnt 0x0" ::: "memory");
}
__device__ __forceinline__ void wait_ld4() {
  asm volatile("s_wait_loadcnt 0x4" ::: "memory");
}

// Plain packed-B load (logits kernel; hoisting is harmless there).
__device__ __forceinline__ v16bf load_b_packed(const __bf16* p) {
  v8bf lo = *(const v8bf*)(p);
  v8bf hi = *(const v8bf*)(p + 8);
  return __builtin_shufflevector(lo, hi, 0, 1, 2, 3, 4, 5, 6, 7,
                                 8, 9, 10, 11, 12, 13, 14, 15);
}

// ---------------------------------------------------------------------------
// Pack a row-major f32 weight W (Ndim x Kdim), used as B[k,n] = scale*W[n,k],
// into the bf16 WMMA B-operand lane layout:
//   dst[((nt*tilesPerNt + tileBase + kt)*32 + lane)*16 + j]
//   lane<16  : K = kt*32 + (j<8 ? j : 16+j-8)     , N = nt*16 + (lane&15)
//   lane>=16 : K = kt*32 + 8 + (j<8 ? j : 16+j-8) , N = nt*16 + (lane&15)
// ---------------------------------------------------------------------------
__global__ void pack_b_kernel(const float* __restrict__ W,
                              __bf16* __restrict__ out,
                              int Ndim, int Kdim,
                              int tilesPerNt, int tileBase, float scale) {
  int tid = blockIdx.x * blockDim.x + threadIdx.x;
  if (tid >= Ndim * Kdim) return;
  int KT   = Kdim >> 5;
  int j    = tid & 15;
  int lane = (tid >> 4) & 31;
  int tile = tid >> 9;
  int kt = tile % KT;
  int nt = tile / KT;
  int n = nt * 16 + (lane & 15);
  int k = kt * 32 + ((j & 8) ? 16 : 0) + ((lane & 16) ? 8 : 0) + (j & 7);
  size_t dst = ((size_t)(nt * tilesPerNt + tileBase + kt) * 32 + lane) * 16 + j;
  out[dst] = (__bf16)(scale * W[(size_t)n * Kdim + k]);
}

// ---------------------------------------------------------------------------
// Cooperative load of one timestep x_t (128x64 f32) into LDS, pre-swizzled
// into the A-operand tile layout.  512 threads, 16 floats (quarter row) each.
// A group of 8 consecutive K values maps to one contiguous 16B chunk:
//   lane_a = (b&15) + 16*((k0>>3)&1),  byte = 16*((k0>>4)&1)
// ---------------------------------------------------------------------------
__device__ __forceinline__ void load_x_tile(const float* __restrict__ xt,
                                            __bf16* xtiles, int tid) {
  int base = tid * 16;                  // 8192 floats / 512 threads
  int b   = base >> 6;
  int i0  = base & 63;                  // 0,16,32,48
  const float4* s = (const float4*)(xt + base);
  float4 p0 = s[0], p1 = s[1], p2 = s[2], p3 = s[3];
  int mt  = b >> 4;
  int bl  = b & 15;
  int kt2 = i0 >> 5;
  int je  = (i0 & 16) >> 1;             // element offset 0 or 8 inside chunk pair
  __bf16* tile = &xtiles[(kt2 * 8 + mt) * TILE_ELEM];
  v8bf qa, qb;
  qa[0] = (__bf16)p0.x; qa[1] = (__bf16)p0.y; qa[2] = (__bf16)p0.z; qa[3] = (__bf16)p0.w;
  qa[4] = (__bf16)p1.x; qa[5] = (__bf16)p1.y; qa[6] = (__bf16)p1.z; qa[7] = (__bf16)p1.w;
  qb[0] = (__bf16)p2.x; qb[1] = (__bf16)p2.y; qb[2] = (__bf16)p2.z; qb[3] = (__bf16)p2.w;
  qb[4] = (__bf16)p3.x; qb[5] = (__bf16)p3.y; qb[6] = (__bf16)p3.z; qb[7] = (__bf16)p3.w;
  *(v8bf*)&tile[bl * 16 + je]        = qa;   // K half 0 -> lanes 0-15
  *(v8bf*)&tile[(bl + 16) * 16 + je] = qb;   // K half 1 -> lanes 16-31
}

// One pipeline stage: A tiles from LDS with ping-pong (issue mt+1's loads
// before mt's WMMAs so ds latency overlaps compute); 2 WMMAs per M-tile.
__device__ __forceinline__ void mm_stage(const __bf16* at,
                                         v4u c0l, v4u c0h, v4u c1l, v4u c1h,
                                         v8f (&a0)[8], v8f (&a1)[8]) {
  v16bf b0 = combine_b(c0l, c0h);
  v16bf b1 = combine_b(c1l, c1h);
  v4u alo = *(const v4u*)(at);
  v4u ahi = *(const v4u*)(at + 8);
#pragma unroll
  for (int mt = 0; mt < 8; ++mt) {
    v4u nlo = alo, nhi = ahi;
    if (mt < 7) {
      nlo = *(const v4u*)(at + (mt + 1) * TILE_ELEM);
      nhi = *(const v4u*)(at + (mt + 1) * TILE_ELEM + 8);
    }
    v16bf am = combine_b(alo, ahi);
    a0[mt] = wmma_bf16(am, b0, a0[mt]);
    a1[mt] = wmma_bf16(am, b1, a1[mt]);
    alo = nlo; ahi = nhi;
  }
}

// ---------------------------------------------------------------------------
// Persistent recurrence kernel: one workgroup, 16 waves.
// Wave w owns N-strips 2w and 2w+1 (32 hidden units) x all 8 M-tiles.
// ---------------------------------------------------------------------------
__global__ __launch_bounds__(512) void ctrnn_recurrent(
    const float* __restrict__ x,        // (T,B,I) f32
    const __bf16* __restrict__ bstream, // (32nt,18 tiles,32,16): aW_in|aW_h
    const float* __restrict__ b_in,
    const float* __restrict__ b_h,
    float* __restrict__ hidden)         // (T,B,H) f32
{
  __shared__ __align__(16) __bf16 frtiles[16 * 8 * TILE_ELEM];  // 128 KB
  __shared__ __align__(16) __bf16 xtiles[2 * 8 * TILE_ELEM];    //  16 KB

  const int tid     = threadIdx.x;
  const int lane    = tid & 31;
  const int wave    = tid >> 5;          // 0..15
  const int nt0     = wave * 2;
  const int laneN   = lane & 15;
  const int halfsel = (lane >> 4) & 1;   // C layout: M = r + 8*halfsel
  const int alane   = lane * 16;         // A-tile per-lane element offset

  // fr_0 = relu(0) = 0
  for (int i = tid; i < 16 * 8 * TILE_ELEM; i += 512) frtiles[i] = (__bf16)0.0f;
  load_x_tile(x, xtiles, tid);
  __syncthreads();

  const int   col0   = nt0 * 16 + laneN;
  const int   col1   = col0 + 16;
  const float absum0 = K_ALPHA * (b_in[col0] + b_h[col0]);
  const float absum1 = K_ALPHA * (b_in[col1] + b_h[col1]);
  const float kDecay = 1.0f - K_ALPHA;   // 0.9

  // fr scatter bases (inverse of the A-operand element mapping) per strip
  auto scatter_base = [&](int col) {
    int kt = col >> 5, kk = col & 31;
    int lane_lo = ((kk >> 3) & 1) << 4;
    int je = ((kk >> 4) & 1) * 8 + (kk & 7);
    return kt * (8 * TILE_ELEM) + (halfsel * 8 + lane_lo) * 16 + je;
  };
  const int fb0 = scatter_base(col0);
  const int fb1 = scatter_base(col1);

  // B stream: ONE pointer; all 36 per-step loads use immediate offsets.
  const __bf16* bp = bstream + ((size_t)nt0 * N_STAGES * 32 + lane) * 16;

  v8f acc0[8], acc1[8];                  // doubles as the v-state
#pragma unroll
  for (int mt = 0; mt < 8; ++mt) { acc0[mt] = v8f{}; acc1[mt] = v8f{}; }

  for (int t = 0; t < T_STEPS; ++t) {
    // ---- decay: acc = (1-a) * v -----------------------------------------
#pragma unroll
    for (int mt = 0; mt < 8; ++mt) { acc0[mt] *= kDecay; acc1[mt] *= kDecay; }

    // warm L2 for next timestep's input while we compute
    if (t + 1 < T_STEPS)
      __builtin_prefetch(x + (size_t)(t + 1) * (BATCH * IN_DIM) + tid * 16, 0, 0);

    // ---- 18-stage B pipeline, double-buffered, immediate offsets --------
    v4u c0l, c0h, c1l, c1h, n0l, n0h, n1l, n1h;
    load_b_imm<0>(bp, c0l, c0h);
    load_b_imm<STRIP_B>(bp, c1l, c1h);
    static_for<N_STAGES>([&](auto sc) {
      constexpr int s = decltype(sc)::value;
      if constexpr (s < N_STAGES - 1) {
        load_b_imm<(s + 1) * 1024>(bp, n0l, n0h);
        load_b_imm<STRIP_B + (s + 1) * 1024>(bp, n1l, n1h);
        wait_ld4();                      // current stage's 4 loads landed
      } else {
        wait_ld0();
      }
      const __bf16* at = (s < 2) ? &xtiles[s * (8 * TILE_ELEM) + alane]
                                 : &frtiles[(s - 2) * (8 * TILE_ELEM) + alane];
      mm_stage(at, c0l, c0h, c1l, c1h, acc0, acc1);
      c0l = n0l; c0h = n0h; c1l = n1l; c1h = n1h;
    });

    __syncthreads();  // everyone done READING frtiles/xtiles for step t

    // ---- v = acc + a*bias; fr = relu(v); publish fr ---------------------
    float* hid_t = hidden + (size_t)t * (BATCH * HID);
#pragma unroll
    for (int mt = 0; mt < 8; ++mt) {
#pragma unroll
      for (int r = 0; r < 8; ++r) {
        float v0 = acc0[mt][r] + absum0;
        float v1 = acc1[mt][r] + absum1;
        acc0[mt][r] = v0;                            // carry v-state
        acc1[mt][r] = v1;
        float f0 = fmaxf(v0, 0.0f);
        float f1 = fmaxf(v1, 0.0f);
        int brow = mt * 16 + r + halfsel * 8;        // C layout row
        hid_t[(size_t)brow * HID + col0] = f0;       // hidden_seq output
        hid_t[(size_t)brow * HID + col1] = f1;
        frtiles[fb0 + mt * TILE_ELEM + r * 16] = (__bf16)f0;  // A-layout scatter
        frtiles[fb1 + mt * TILE_ELEM + r * 16] = (__bf16)f1;
      }
    }
    if (t + 1 < T_STEPS)
      load_x_tile(x + (size_t)(t + 1) * (BATCH * IN_DIM), xtiles, tid);

    __syncthreads();  // fr/x for step t+1 visible before next WMMA phase
  }
}

// ---------------------------------------------------------------------------
// logits = hidden @ W_out^T + b_out.  M = T*B = 65536 rows, N = 32, K = 512.
// 8 waves/block, one 16-row M-tile per wave, both N-tiles accumulated.
// Memory-bound: 134 MB hidden read once, coalesced 16B loads.
// ---------------------------------------------------------------------------
__global__ __launch_bounds__(256) void ctrnn_logits(
    const float* __restrict__ hidden,   // (T*B, H) f32
    const __bf16* __restrict__ pWout,   // packed B layout (2nt,16kt,32,16)
    const float* __restrict__ b_out,
    float* __restrict__ logits)         // (T*B, O) f32
{
  const int lane    = threadIdx.x & 31;
  const int wave    = threadIdx.x >> 5;
  const int mtile   = blockIdx.x * 8 + wave;      // 0..4095
  const int laneN   = lane & 15;
  const int halfsel = (lane >> 4) & 1;

  const float* arow = hidden + (size_t)mtile * 16 * HID + (size_t)laneN * HID;

  v8f acc0 = v8f{};
  v8f acc1 = v8f{};

  for (int kt = 0; kt < 16; ++kt) {
    const int kb = kt * 32 + halfsel * 8;
    const float* pa = arow + kb;
    v4f f0 = *(const v4f*)(pa);
    v4f f1 = *(const v4f*)(pa + 4);
    v4f f2 = *(const v4f*)(pa + 16);
    v4f f3 = *(const v4f*)(pa + 20);
    v16bf am;
#pragma unroll
    for (int j = 0; j < 4; ++j) {
      am[j]      = (__bf16)f0[j];
      am[j + 4]  = (__bf16)f1[j];
      am[j + 8]  = (__bf16)f2[j];
      am[j + 12] = (__bf16)f3[j];
    }
    v16bf b0 = load_b_packed(&pWout[((size_t)(0 * 16 + kt) * 32 + lane) * 16]);
    v16bf b1 = load_b_packed(&pWout[((size_t)(1 * 16 + kt) * 32 + lane) * 16]);
    acc0 = wmma_bf16(am, b0, acc0);
    acc1 = wmma_bf16(am, b1, acc1);
  }

  const float bo0 = b_out[laneN];
  const float bo1 = b_out[16 + laneN];
#pragma unroll
  for (int r = 0; r < 8; ++r) {
    size_t row = (size_t)mtile * 16 + r + halfsel * 8;
    logits[row * OUT_DIM + laneN]      = acc0[r] + bo0;
    logits[row * OUT_DIM + 16 + laneN] = acc1[r] + bo1;
  }
}

// ---------------------------------------------------------------------------
// Launch: pack weights (bf16 B-layout, alpha folded in; W_in+W_h interleaved
// into one per-strip B stream) -> persistent recurrence -> logits.
// d_in order: inputs, W_in, b_in, W_h, b_h, W_out, b_out (all f32).
// d_out: logits (T*B*O) then hidden_seq (T*B*H), f32.
// d_ws: ~620 KB of packed bf16 weights.
// ---------------------------------------------------------------------------
extern "C" void kernel_launch(void* const* d_in, const int* in_sizes, int n_in,
                              void* d_out, int out_size, void* d_ws, size_t ws_size,
                              hipStream_t stream) {
  const float* inputs = (const float*)d_in[0];
  const float* W_in   = (const float*)d_in[1];
  const float* b_in   = (const float*)d_in[2];
  const float* W_h    = (const float*)d_in[3];
  const float* b_h    = (const float*)d_in[4];
  const float* W_out  = (const float*)d_in[5];
  const float* b_out  = (const float*)d_in[6];

  float* logits = (float*)d_out;
  float* hidden = logits + (size_t)T_STEPS * BATCH * OUT_DIM;

  __bf16* bstream = (__bf16*)d_ws;                    // 32*18*512 elems
  __bf16* pWout   = bstream + (size_t)32 * STRIP;     // 32*512 elems

  // W_in -> stream tiles 0..1, W_h -> stream tiles 2..17 (per N-strip)
  pack_b_kernel<<<(HID * IN_DIM + 255) / 256, 256, 0, stream>>>(
      W_in, bstream, HID, IN_DIM, N_STAGES, 0, K_ALPHA);
  pack_b_kernel<<<(HID * HID + 255) / 256, 256, 0, stream>>>(
      W_h, bstream, HID, HID, N_STAGES, 2, K_ALPHA);
  pack_b_kernel<<<(OUT_DIM * HID + 255) / 256, 256, 0, stream>>>(
      W_out, pWout, OUT_DIM, HID, 16, 0, 1.0f);

  ctrnn_recurrent<<<1, 512, 0, stream>>>(inputs, bstream, b_in, b_h, hidden);

  ctrnn_logits<<<(T_STEPS * BATCH / 16) / 8, 256, 0, stream>>>(
      hidden, pWout, b_out, logits);
}
